// HexConv_83897891160712
// MI455X (gfx1250) — matrix operand order repaired and checked by reading, not compiled, not run on hardware
//
#include <hip/hip_runtime.h>

typedef _Float16 h8   __attribute__((ext_vector_type(8)));
typedef _Float16 v16h __attribute__((ext_vector_type(16)));
typedef float    v8f  __attribute__((ext_vector_type(8)));
typedef float    f4   __attribute__((ext_vector_type(4)));
typedef unsigned int u32x4 __attribute__((ext_vector_type(4)));
typedef int          i32x4 __attribute__((ext_vector_type(4)));
typedef int          i32x8 __attribute__((ext_vector_type(8)));

#define HEXES   165
#define CH      128      // input channels per hex (= K per neighbor chunk)
#define OUTC    128      // output channels
#define KTOT    896      // 7 * 128
#define MT      128      // M rows per block
#define APAD    136      // 128 + 8 halfwords pad (272B row stride, conflict-free)
#define A_HALFS (MT * APAD)      // 17408 halfwords
#define B_HALFS (OUTC * APAD)    // 17408 halfwords per W-chunk buffer
#define A_BYTES (A_HALFS * 2)    // 34816
#define B_BYTES (B_HALFS * 2)    // 34816

#if __has_builtin(__builtin_amdgcn_tensor_load_to_lds) && \
    __has_builtin(__builtin_amdgcn_s_wait_tensorcnt)
#define USE_TDM 1
#else
#define USE_TDM 0
#endif

// ---------------------------------------------------------------------------
// Pass 1: W [896,128] f32 (k-major) -> Wt [128,896] f16 (o-major, k contiguous)
// ---------------------------------------------------------------------------
__global__ __launch_bounds__(256) void hexconv_wt_kernel(
    const float* __restrict__ W, _Float16* __restrict__ Wt) {
  int idx = blockIdx.x * 256 + threadIdx.x;   // 0 .. 128*896-1
  int o = idx / KTOT;
  int k = idx - o * KTOT;
  Wt[idx] = (_Float16)W[k * OUTC + o];
}

// ---------------------------------------------------------------------------
// Pass 2: gathered GEMM with v_wmma_f32_16x16x32_f16 + TDM-staged W chunks
// ---------------------------------------------------------------------------
__global__ __launch_bounds__(256) void hexconv_main_kernel(
    const float* __restrict__ x, const _Float16* __restrict__ Wt,
    const float* __restrict__ bias, float* __restrict__ out) {
  extern __shared__ _Float16 smem[];
  _Float16* As = smem;                    // [MT][APAD] A tile (single buffer)
  // Bs double buffer: smem + A_HALFS + buf*B_HALFS, [OUTC][APAD] each

  const int tid   = threadIdx.x;
  const int wave  = tid >> 5;
  const int lane  = tid & 31;
  const int mbase = blockIdx.x * MT;

  // ---- per-thread A-gather assignment: 2 threads per row, 64 cols each ----
  const int ar = tid >> 1;             // row in tile: 0..127 (doubles as o-row)
  const int ac = (tid & 1) * 64;       // col base: 0 or 64
  const int gm = mbase + ar;           // global row
  const int bb = gm / HEXES;           // batch
  const int n  = gm - bb * HEXES;      // hex index 0..164
  const int y  = n / 15 + 1;           // padded-grid row 1..11
  const int xx = n - (y - 1) * 15 + 1; // padded-grid col 1..15
  const int pbase = y * 17 + xx;
  const bool yeven = (y & 1) == 0;
  const int off0[7] = {-17, -16, -1, 0, 1, 17, 18};
  const int off1[7] = {-18, -17, -1, 0, 1, 16, 17};

  // ---- WMMA fragment addressing (per ISA VGPR layout tables) ----
  const int arow = wave * 16 + (lane & 15);
  const int sel  = lane >> 4;

  const unsigned ldsBase = (unsigned)(size_t)(void*)smem;  // LDS byte offset

  // ---- stage next W chunk (chunk j) into Bs[buf] ----
  auto stageW = [&](int j, int buf) {
#if USE_TDM
    if (wave == 0) {
      unsigned long long ga = (unsigned long long)(size_t)(Wt + j * CH);
      unsigned ldsOff = ldsBase + (unsigned)A_BYTES + (unsigned)buf * B_BYTES;
      u32x4 g0;
      g0[0] = 1u;                                   // count=1 (valid user D#)
      g0[1] = ldsOff;                               // lds_addr
      g0[2] = (unsigned)(ga & 0xffffffffu);         // global_addr[31:0]
      g0[3] = (unsigned)((ga >> 32) & 0x01ffffffu)  // global_addr[56:32]
              | (2u << 30);                         // type = 2 ("image")
      i32x8 g1;
      g1[0] = (1 << 16)      // data_size = 1 (2 bytes)
            | (1 << 20)      // pad_enable
            | (5 << 22)      // pad_interval: 64 dwords (one 128-f16 row)
            | (3 << 25);     // pad_amount: 4 dwords (8 f16) -> 272B stride
      g1[1] = (KTOT << 16);              // tensor_dim0 = 896 (lo 16)
      g1[2] = (OUTC << 16);              // tensor_dim1 = 128 (lo 16)
      g1[3] = (CH << 16);                // tile_dim0 = 128
      g1[4] = OUTC;                      // tile_dim1 = 128, tile_dim2 = 0
      g1[5] = KTOT;                      // tensor_dim0_stride = 896
      g1[6] = 0;
      g1[7] = 0;
      i32x4 gz4 = {0, 0, 0, 0};
      i32x8 gz8 = {0, 0, 0, 0, 0, 0, 0, 0};
      __builtin_amdgcn_tensor_load_to_lds(g0, g1, gz4, gz4, gz8, 0);
    }
#else
    _Float16* dst = smem + A_HALFS + buf * B_HALFS + ar * APAD + ac;
    const _Float16* ws = Wt + ar * KTOT + j * CH + ac;
#pragma unroll
    for (int c = 0; c < 64; c += 8)
      *(h8*)(dst + c) = *(const h8*)(ws + c);
#endif
  };
  auto waitW = [&]() {
#if USE_TDM
    if (wave == 0) __builtin_amdgcn_s_wait_tensorcnt(0);
#endif
  };

  // ---- register-staged A gather (global f32 -> f16 regs) ----
  h8 aStage[8];
  auto loadA = [&](int j) {
    int p  = pbase + (yeven ? off0[j] : off1[j]);
    int py = p / 17;
    int px = p - py * 17;
    bool valid = (py >= 1) && (py <= 11) && (px >= 1) && (px <= 15);
    int src = (py - 1) * 15 + (px - 1);
    const float* xs = x + ((bb * HEXES + src) * CH + ac);
    if (valid) {
#pragma unroll
      for (int c = 0; c < 8; ++c) {
        f4 v0 = *(const f4*)(xs + c * 8);
        f4 v1 = *(const f4*)(xs + c * 8 + 4);
        aStage[c] = h8{(_Float16)v0[0], (_Float16)v0[1], (_Float16)v0[2],
                       (_Float16)v0[3], (_Float16)v1[0], (_Float16)v1[1],
                       (_Float16)v1[2], (_Float16)v1[3]};
      }
    } else {
      h8 z = {};
#pragma unroll
      for (int c = 0; c < 8; ++c) aStage[c] = z;
    }
  };
  auto storeA = [&]() {
    _Float16* dst = As + ar * APAD + ac;
#pragma unroll
    for (int c = 0; c < 8; ++c) *(h8*)(dst + c * 8) = aStage[c];
  };

  v8f acc[8] = {};

  // -------- prologue: stage chunk 0 --------
  stageW(0, 0);
  loadA(0);
  storeA();
  waitW();
  __syncthreads();

  // -------- main pipeline over 7 neighbor chunks --------
  for (int j = 0; j < 7; ++j) {
    // issue next chunk's staging first so it overlaps this chunk's WMMAs
    if (j < 6) {
      stageW(j + 1, (j + 1) & 1);   // TDM into the other B buffer
      loadA(j + 1);                 // global loads in flight during compute
    }

    // -------- compute chunk j: 4 K-steps x 8 N-tiles --------
    const _Float16* Bsj = smem + A_HALFS + (j & 1) * B_HALFS;
#pragma unroll
    for (int ks = 0; ks < CH; ks += 32) {
      h8 a_lo = *(const h8*)(As + arow * APAD + ks + sel * 8);
      h8 a_hi = *(const h8*)(As + arow * APAD + ks + 16 + sel * 8);
      v16h a = __builtin_shufflevector(a_lo, a_hi, 0, 1, 2, 3, 4, 5, 6, 7, 8,
                                       9, 10, 11, 12, 13, 14, 15);
      v16h bm[8];
#pragma unroll
      for (int nt = 0; nt < 8; ++nt) {
        const _Float16* bp = Bsj + (nt * 16 + (lane & 15)) * APAD + ks + sel * 8;
        h8 b_lo = *(const h8*)(bp);
        h8 b_hi = *(const h8*)(bp + 16);
        bm[nt] = __builtin_shufflevector(b_lo, b_hi, 0, 1, 2, 3, 4, 5, 6, 7, 8,
                                         9, 10, 11, 12, 13, 14, 15);
      }
#pragma unroll
      for (int nt = 0; nt < 8; ++nt) {
        acc[nt] = __builtin_amdgcn_wmma_f32_16x16x32_f16(
            false, a, false, bm[nt], (short)0, acc[nt], false, false);
      }
    }

    if (j < 6) {
      __syncthreads();   // all waves done reading As for chunk j
      storeA();          // write A(j+1)
      waitW();           // W(j+1) TDM complete (wave 0's TENSORcnt)
      __syncthreads();   // As + Bs[(j+1)&1] visible to all waves
    }
  }

  // -------- epilogue: C/D layout -> global, add bias --------
  const int mrow0 = mbase + wave * 16 + (lane >> 4) * 8;
  const int ncol  = lane & 15;
#pragma unroll
  for (int nt = 0; nt < 8; ++nt) {
    int o = nt * 16 + ncol;
    float bv = bias[o];
#pragma unroll
    for (int c = 0; c < 8; ++c) {
      out[(mrow0 + c) * OUTC + o] = acc[nt][c] + bv;
    }
  }
}

// ---------------------------------------------------------------------------
extern "C" void kernel_launch(void* const* d_in, const int* in_sizes, int n_in,
                              void* d_out, int out_size, void* d_ws,
                              size_t ws_size, hipStream_t stream) {
  const float* x    = (const float*)d_in[0];  // [1024,165,128] f32
  const float* W    = (const float*)d_in[1];  // [896,128] f32
  const float* bias = (const float*)d_in[2];  // [128] f32
  float* out        = (float*)d_out;          // [1024,165,128] f32
  _Float16* Wt      = (_Float16*)d_ws;        // [128,896] f16 (229376 B)

  hexconv_wt_kernel<<<(OUTC * KTOT) / 256, 256, 0, stream>>>(W, Wt);

  const int mtotal = 1024 * HEXES;            // 168960, divisible by 128
  const size_t lds = (size_t)(A_BYTES + 2 * B_BYTES);  // 104448 B
  hexconv_main_kernel<<<mtotal / MT, 256, lds, stream>>>(x, Wt, bias, out);
}
